// GroupedExpertsExpertChoice_18451179504169
// MI455X (gfx1250) — matrix operand order repaired and checked by reading, not compiled, not run on hardware
//
#include <hip/hip_runtime.h>
#include <hip/hip_bf16.h>

typedef __bf16 bf16;
typedef __attribute__((ext_vector_type(16))) __bf16 v16bf;
typedef __attribute__((ext_vector_type(8)))  __bf16 v8bf;
typedef __attribute__((ext_vector_type(4)))  __bf16 v4bf;
typedef __attribute__((ext_vector_type(8)))  float  v8f;
typedef __attribute__((ext_vector_type(4)))  float  v4f;

#define Bv 4
#define Sv 4096
#define Dv 1024
#define Ev 8
#define Cv 1024
#define Fv 2048

// ---------------------------------------------------------------------------
// CDNA5 async global->LDS copy (GLOBAL_LOAD_ASYNC_TO_LDS_B128, ASYNCcnt).
// LDS destination = wave-relative LDS byte address = low 32 bits of the flat
// shared-aperture pointer (ISA 10.2).
// ---------------------------------------------------------------------------
__device__ __forceinline__ void async_load_b128(const void* gptr, void* lptr) {
  unsigned l = (unsigned)(uintptr_t)lptr;
  asm volatile("global_load_async_to_lds_b128 %0, %1, off"
               :: "v"(l), "v"(gptr) : "memory");
}
__device__ __forceinline__ void wait_async0() {
  asm volatile("s_wait_asynccnt 0" ::: "memory");
}

__device__ __forceinline__ float fast_sigmoid(float x) {
  return __builtin_amdgcn_rcpf(1.0f + __expf(-x));   // v_rcp_f32 + v_exp_f32
}

// ---------------------------------------------------------------------------
// WMMA fragment loaders, matching the CDNA5 ISA VGPR layouts (CDNA5 ISA).
// ---------------------------------------------------------------------------
__device__ __forceinline__ v16bf ldsA_frag(const bf16* base, int rowBase) {
  const int lane = threadIdx.x & 31;
  const int m = lane & 15, h = lane >> 4;
  const bf16* p = base + (rowBase + m) * 32;
  v8bf lo = *(const v8bf*)(p + 8 * h);
  v8bf hi = *(const v8bf*)(p + 16 + 8 * h);
  v16bf r;
#pragma unroll
  for (int j = 0; j < 8; ++j) { r[j] = lo[j]; r[j + 8] = hi[j]; }
  return r;
}

__device__ __forceinline__ v16bf ldsB_frag(const bf16* base, int nBase) {
  const int lane = threadIdx.x & 31;
  const int n = lane & 15, h = lane >> 4;
  const bf16* p = base + (nBase + n) * 32;
  v8bf lo = *(const v8bf*)(p + 8 * h);
  v8bf hi = *(const v8bf*)(p + 16 + 8 * h);
  v16bf r;
#pragma unroll
  for (int j = 0; j < 8; ++j) { r[j] = lo[j]; r[j + 8] = hi[j]; }
  return r;
}

// ---------------------------------------------------------------------------
// Tiled transpose + f32->bf16 convert:  in[e][r][c] (RxCc) -> out[e][c][r]
// ---------------------------------------------------------------------------
__global__ __launch_bounds__(256) void transconv(const float* __restrict__ in,
                                                 bf16* __restrict__ out,
                                                 int R, int Cc) {
  __shared__ __align__(16) float tile[32][33];
  const int e = blockIdx.z;
  const int col = blockIdx.x * 32;
  const int row = blockIdx.y * 32;
  const int tx = threadIdx.x & 31;
  const int ty = threadIdx.x >> 5;  // 0..7
  const float* src = in + (size_t)e * R * Cc;
  bf16* dst = out + (size_t)e * R * Cc;
#pragma unroll
  for (int i = 0; i < 4; ++i)
    tile[ty + 8 * i][tx] = src[(size_t)(row + ty + 8 * i) * Cc + col + tx];
  __syncthreads();
#pragma unroll
  for (int i = 0; i < 4; ++i)
    dst[(size_t)(col + ty + 8 * i) * R + row + tx] = (bf16)tile[tx][ty + 8 * i];
}

__global__ void zero_f32(float4* __restrict__ p, long long n4) {
  long long i = (long long)blockIdx.x * blockDim.x + threadIdx.x;
  long long stride = (long long)gridDim.x * blockDim.x;
  float4 z = make_float4(0.f, 0.f, 0.f, 0.f);
  for (; i < n4; i += stride) p[i] = z;
}

// ---------------------------------------------------------------------------
// Kernel A: gathered SwiGLU up-projection, h = silu(X*W1) * (X*W2) -> bf16
// ---------------------------------------------------------------------------
__global__ __launch_bounds__(256) void moe_gate_value(
    const float* __restrict__ x, const long long* __restrict__ tok,
    const bf16* __restrict__ w1T, const bf16* __restrict__ w2T,
    bf16* __restrict__ h, int b) {
  __shared__ __align__(16) bf16 lA[128 * 32];
  __shared__ __align__(16) bf16 lB1[64 * 32];
  __shared__ __align__(16) bf16 lB2[64 * 32];
  __shared__ int rowIdx[128];

  const int e = blockIdx.z;
  const int cBase = blockIdx.y * 128;
  const int fBase = blockIdx.x * 64;
  const int t = threadIdx.x;
  const int wave = t >> 5, lane = t & 31;
  const int wm = wave >> 1, wn = wave & 1;

  if (t < 128)
    rowIdx[t] = (int)tok[((size_t)b * Ev + e) * Cv + cBase + t];
  __syncthreads();

  v8f accG[2][2], accV[2][2];
#pragma unroll
  for (int i = 0; i < 2; ++i)
#pragma unroll
    for (int j = 0; j < 2; ++j) { accG[i][j] = (v8f)0.f; accV[i][j] = (v8f)0.f; }

  const float* xb = x + (size_t)b * Sv * Dv;
  const bf16* w1e = w1T + ((size_t)e * Fv + fBase) * Dv;
  const bf16* w2e = w2T + ((size_t)e * Fv + fBase) * Dv;

  // hoisted gather row pointers + LDS store slots
  const int gr = t >> 3;          // 0..31
  const int gk = (t & 7) * 4;     // 0..28
  const float* rowPtr[4];
  bf16* ldsSlot[4];
#pragma unroll
  for (int p = 0; p < 4; ++p) {
    rowPtr[p] = xb + (size_t)rowIdx[p * 32 + gr] * Dv + gk;
    ldsSlot[p] = lA + (p * 32 + gr) * 32 + gk;
  }
  // hoisted async weight-tile addresses
  const int bn = t >> 2;          // 0..63
  const int bk = (t & 3) * 8;     // 0..24
  const bf16* w1p = w1e + (size_t)bn * Dv + bk;
  const bf16* w2p = w2e + (size_t)bn * Dv + bk;
  bf16* lB1p = lB1 + bn * 32 + bk;
  bf16* lB2p = lB2 + bn * 32 + bk;

  for (int k0 = 0; k0 < Dv; k0 += 32) {
    // weight tiles: async DMA straight into LDS (already bf16, [n][k] order)
    async_load_b128(w1p + k0, lB1p);
    async_load_b128(w2p + k0, lB2p);
    // gathered x tile: load f32, convert, ds-store
#pragma unroll
    for (int p = 0; p < 4; ++p) {
      v4f v = *(const v4f*)(rowPtr[p] + k0);
      v4bf o;
      o[0] = (bf16)v[0]; o[1] = (bf16)v[1]; o[2] = (bf16)v[2]; o[3] = (bf16)v[3];
      *(v4bf*)ldsSlot[p] = o;
    }
    wait_async0();
    __syncthreads();

    v16bf fa[2], fb1[2], fb2[2];
#pragma unroll
    for (int i = 0; i < 2; ++i) fa[i] = ldsA_frag(lA, wm * 32 + i * 16);
#pragma unroll
    for (int j = 0; j < 2; ++j) {
      fb1[j] = ldsB_frag(lB1, wn * 32 + j * 16);
      fb2[j] = ldsB_frag(lB2, wn * 32 + j * 16);
    }
#pragma unroll
    for (int i = 0; i < 2; ++i)
#pragma unroll
      for (int j = 0; j < 2; ++j) {
        accG[i][j] = __builtin_amdgcn_wmma_f32_16x16x32_bf16(
            false, fa[i], false, fb1[j], (short)0, accG[i][j], false, false);
        accV[i][j] = __builtin_amdgcn_wmma_f32_16x16x32_bf16(
            false, fa[i], false, fb2[j], (short)0, accV[i][j], false, false);
      }
    __syncthreads();
  }

  // epilogue: h = silu(G)*V (C/D layout: m = r + 8*(lane/16), n = lane%16)
  const int n = lane & 15, hh = lane >> 4;
  bf16* hb = h + (size_t)e * Cv * Fv;
#pragma unroll
  for (int i = 0; i < 2; ++i)
#pragma unroll
    for (int j = 0; j < 2; ++j)
#pragma unroll
      for (int r = 0; r < 8; ++r) {
        const int c = cBase + wm * 32 + i * 16 + r + 8 * hh;
        const int f = fBase + wn * 32 + j * 16 + n;
        const float g = accG[i][j][r];
        const float vv = accV[i][j][r];
        hb[(size_t)c * Fv + f] = (bf16)(g * fast_sigmoid(g) * vv);
      }
}

// ---------------------------------------------------------------------------
// Kernel B: out_tokens = h @ W3, scale by expert_weights, atomic scatter-add
// ---------------------------------------------------------------------------
__global__ __launch_bounds__(256) void moe_down_scatter(
    const bf16* __restrict__ h, const bf16* __restrict__ w3T,
    const long long* __restrict__ tok, const float* __restrict__ ew,
    float* __restrict__ out, int b) {
  __shared__ __align__(16) bf16 lA[128 * 32];
  __shared__ __align__(16) bf16 lB[64 * 32];
  __shared__ int rowIdx[128];
  __shared__ float scale[128];

  const int e = blockIdx.z;
  const int cBase = blockIdx.y * 128;
  const int dBase = blockIdx.x * 64;
  const int t = threadIdx.x;
  const int wave = t >> 5, lane = t & 31;
  const int wm = wave >> 1, wn = wave & 1;

  if (t < 128) {
    rowIdx[t] = (int)tok[((size_t)b * Ev + e) * Cv + cBase + t];
    scale[t] = ew[((size_t)b * Ev + e) * Cv + cBase + t];
  }
  __syncthreads();

  v8f acc[2][2];
#pragma unroll
  for (int i = 0; i < 2; ++i)
#pragma unroll
    for (int j = 0; j < 2; ++j) acc[i][j] = (v8f)0.f;

  const bf16* ha = h + (size_t)e * Cv * Fv + (size_t)cBase * Fv;
  const bf16* w3e = w3T + ((size_t)e * Dv + dBase) * Fv;

  // hoisted async source/dest addresses (everything is bf16 already)
  const int ar = t >> 1;           // 0..127
  const int ao = (t & 1) * 16;     // 0 or 16
  const bf16* hap = ha + (size_t)ar * Fv + ao;
  bf16* lAp = lA + ar * 32 + ao;
  const int bn = t >> 2;           // 0..63
  const int bk = (t & 3) * 8;      // 0..24
  const bf16* w3p = w3e + (size_t)bn * Fv + bk;
  bf16* lBp = lB + bn * 32 + bk;

  for (int k0 = 0; k0 < Fv; k0 += 32) {
    async_load_b128(hap + k0, lAp);
    async_load_b128(hap + k0 + 8, lAp + 8);
    async_load_b128(w3p + k0, lBp);
    wait_async0();
    __syncthreads();

    v16bf fa[2], fb[2];
#pragma unroll
    for (int i = 0; i < 2; ++i) fa[i] = ldsA_frag(lA, wm * 32 + i * 16);
#pragma unroll
    for (int j = 0; j < 2; ++j) fb[j] = ldsB_frag(lB, wn * 32 + j * 16);
#pragma unroll
    for (int i = 0; i < 2; ++i)
#pragma unroll
      for (int j = 0; j < 2; ++j)
        acc[i][j] = __builtin_amdgcn_wmma_f32_16x16x32_bf16(
            false, fa[i], false, fb[j], (short)0, acc[i][j], false, false);
    __syncthreads();
  }

  const int n = lane & 15, hh = lane >> 4;
#pragma unroll
  for (int i = 0; i < 2; ++i)
#pragma unroll
    for (int j = 0; j < 2; ++j)
#pragma unroll
      for (int r = 0; r < 8; ++r) {
        const int cl = wm * 32 + i * 16 + r + 8 * hh;      // local token row
        const int d = dBase + wn * 32 + j * 16 + n;
        const float v = acc[i][j][r] * scale[cl];
        float* dst = out + ((size_t)b * Sv + rowIdx[cl]) * Dv + d;
        unsafeAtomicAdd(dst, v);                            // global_atomic_add_f32
      }
}

// ---------------------------------------------------------------------------
extern "C" void kernel_launch(void* const* d_in, const int* in_sizes, int n_in,
                              void* d_out, int out_size, void* d_ws, size_t ws_size,
                              hipStream_t stream) {
  const float* x       = (const float*)d_in[0];
  const float* ew      = (const float*)d_in[1];
  const long long* tok = (const long long*)d_in[2];
  const float* w1      = (const float*)d_in[3];
  const float* w2      = (const float*)d_in[4];
  const float* w3      = (const float*)d_in[5];
  float* out = (float*)d_out;

  char* ws = (char*)d_ws;
  const size_t wbytes = (size_t)Ev * Dv * Fv * sizeof(bf16);  // 32 MB
  bf16* w1T  = (bf16*)(ws);
  bf16* w2T  = (bf16*)(ws + wbytes);
  bf16* w3T  = (bf16*)(ws + 2 * wbytes);
  bf16* hbuf = (bf16*)(ws + 3 * wbytes);                      // E*C*F bf16, per-b

  zero_f32<<<2048, 256, 0, stream>>>((float4*)out, (long long)Bv * Sv * Dv / 4);

  dim3 tc12(Fv / 32, Dv / 32, Ev);
  transconv<<<tc12, 256, 0, stream>>>(w1, w1T, Dv, Fv);   // -> [e][f][d]
  transconv<<<tc12, 256, 0, stream>>>(w2, w2T, Dv, Fv);   // -> [e][f][d]
  dim3 tc3(Dv / 32, Fv / 32, Ev);
  transconv<<<tc3, 256, 0, stream>>>(w3, w3T, Fv, Dv);    // -> [e][d][f]

  dim3 gA(Fv / 64, Cv / 128, Ev);
  dim3 gB(Dv / 64, Cv / 128, Ev);
  for (int b = 0; b < Bv; ++b) {
    moe_gate_value<<<gA, 256, 0, stream>>>(x, tok, w1T, w2T, hbuf, b);
    moe_down_scatter<<<gB, 256, 0, stream>>>(hbuf, w3T, tok, ew, out, b);
  }
}
